// ExDenoiseFlow_28372553957671
// MI455X (gfx1250) — matrix operand order repaired and checked by reading, not compiled, not run on hardware
//
#include <hip/hip_runtime.h>
#include <hip/hip_bf16.h>
#include <math.h>

typedef __attribute__((ext_vector_type(16))) _Float16 v16h;
typedef __attribute__((ext_vector_type(8)))  float    v8f;

#define BN     16384      // B*N rows
#define NPTS   2048
#define NB     8
#define CLAMP_C 1.9f
#define LOG2PI 1.8378770664093453f

#define ACT_NONE  0
#define ACT_LRELU 1
#define ACT_RELU  2

__device__ __forceinline__ float soft_clamp_f(float s){
  return (2.0f*CLAMP_C/3.14159265358979323846f)*atanf(s/CLAMP_C);
}
__device__ __forceinline__ float act_apply(float v, int act){
  if (act==ACT_LRELU) return v>=0.f? v : 0.01f*v;
  if (act==ACT_RELU)  return v>0.f? v : 0.f;
  return v;
}

// ---------------------------------------------------------------------------
// Generic fused linear layer: Y = act( [X0|X1] @ W + b )
// X0: c0 channels (row stride ld0); X1: I-c0 channels (row stride ld1, may be null)
// W: [I,O] row-major f32 -> f16 in LDS, stored COLUMN-major (wlds[o*96+k]) so a
// lane's B fragment (K = kk+khalf*8+e and kk+16+khalf*8+e, fixed col n) is two contiguous 16B runs
// (2 x ds_load_b128). A fragment = two contiguous 8-float runs per lane.
// SIMPLE=true: single source, I%32==0, 16B-aligned rows -> branch-free K loop
// (2 x float4 per run, unconditional). SIMPLE=false: guarded fast/slow paths.
// One block = 8 waves x 16 rows = 128 rows; M must be a multiple of 128.
// NT = number of 16-wide output tiles (O <= NT*16), compile-time.
// ---------------------------------------------------------------------------
template<int NT, bool SIMPLE>
__global__ __launch_bounds__(256)
void lin_wmma_kernel(const float* __restrict__ X0, int c0, int ld0,
                     const float* __restrict__ X1, int ld1,
                     const float* __restrict__ W, const float* __restrict__ Bb,
                     float* __restrict__ Y, int ldy,
                     int M, int I, int O, int act)
{
  __shared__ _Float16 wlds[64*96];   // [o][k], k padded to KP (<=96), zeros beyond
  __shared__ float    blds[64];
  const int tid = threadIdx.x;
  const int KP  = (I + 31) & ~31;    // <= 96

  for (int idx = tid; idx < KP*64; idx += 256){
    int k = idx >> 6, o = idx & 63;
    float v = (k < I && o < O) ? W[(size_t)k*O + o] : 0.f;
    wlds[o*96 + k] = (_Float16)v;
  }
  if (tid < 64) blds[tid] = (tid < O) ? Bb[tid] : 0.f;
  __syncthreads();

  const int lane   = tid & 31;
  const int wave   = tid >> 5;
  const int rowBase= blockIdx.x*128 + wave*16;
  const int khalf  = lane >> 4;
  int ar = rowBase + (lane & 15);
  if (ar > M-1) ar = M-1;            // clamp (results discarded at store)

  const float* row0 = X0 + (size_t)ar*ld0;
  const float* row1 = X1 ? (X1 + (size_t)ar*ld1) : nullptr;
  __builtin_prefetch(row0, 0, 1);    // global_prefetch_b8

  const bool al0 = ((ld0 & 3) == 0); // row 16B-aligned -> b128 loads legal
  const bool al1 = ((ld1 & 3) == 0);

  v8f acc[NT] = {};

  for (int kk = 0; kk < KP; kk += 32){
    // A fragment (ISA 7.12.2): lane covers K runs [k0,k0+8) for run=0,1
    v16h af;
    #pragma unroll
    for (int run = 0; run < 2; ++run){
      const int k0 = kk + run*16 + khalf*8;
      float rv[8];
      if (SIMPLE){
        // branch-free: always in-bounds (I multiple of 32, single source)
        float4 a = *(const float4*)(row0 + k0);
        float4 b = *(const float4*)(row0 + k0 + 4);
        rv[0]=a.x; rv[1]=a.y; rv[2]=a.z; rv[3]=a.w;
        rv[4]=b.x; rv[5]=b.y; rv[6]=b.z; rv[7]=b.w;
      } else if (al0 && (k0 + 8 <= c0)){
        float4 a = *(const float4*)(row0 + k0);
        float4 b = *(const float4*)(row0 + k0 + 4);
        rv[0]=a.x; rv[1]=a.y; rv[2]=a.z; rv[3]=a.w;
        rv[4]=b.x; rv[5]=b.y; rv[6]=b.z; rv[7]=b.w;
      } else if (al1 && row1 && (k0 >= c0) && (k0 + 8 <= I)){
        float4 a = *(const float4*)(row1 + (k0 - c0));
        float4 b = *(const float4*)(row1 + (k0 - c0) + 4);
        rv[0]=a.x; rv[1]=a.y; rv[2]=a.z; rv[3]=a.w;
        rv[4]=b.x; rv[5]=b.y; rv[6]=b.z; rv[7]=b.w;
      } else {
        #pragma unroll
        for (int e = 0; e < 8; ++e){
          int k = k0 + e;
          float v = 0.f;
          if (k < I) v = (k < c0) ? row0[k] : row1[k - c0];
          rv[e] = v;
        }
      }
      #pragma unroll
      for (int e = 0; e < 8; ++e) af[run*8 + e] = (_Float16)rv[e];
    }
    // B fragments: two 16B runs per lane (K = kk+khalf*8+e, kk+16+khalf*8+e) from column-major LDS
    #pragma unroll
    for (int t = 0; t < NT; ++t){
      const int n = (t<<4) + (lane & 15);
      const _Float16* bp = &wlds[n*96 + kk + (khalf<<3)];
      v16h bf;
      #pragma unroll
      for (int e = 0; e < 8; ++e){ bf[e] = bp[e]; bf[8 + e] = bp[16 + e]; }
      acc[t] = __builtin_amdgcn_wmma_f32_16x16x32_f16(
          false, af, false, bf, (short)0, acc[t], false, false);
    }
  }

  // C/D layout: VGPR v -> row = 8*(lane>=16)+v ; col = lane%16
  const int mg = (lane >> 4) * 8;
  const int nc = lane & 15;
  #pragma unroll
  for (int t = 0; t < NT; ++t){
    int col = (t<<4) + nc;
    if (col >= O) continue;
    #pragma unroll
    for (int v = 0; v < 8; ++v){
      int row = rowBase + mg + v;
      if (row < M){
        float val = acc[t][v] + blds[col];
        Y[(size_t)row*ldy + col] = act_apply(val, act);
      }
    }
  }
}

// ---------------------------------------------------------------------------
// kNN: one thread per query point; candidates tiled through LDS.
// ---------------------------------------------------------------------------
__global__ __launch_bounds__(256)
void knn_kernel(const float* __restrict__ F, int C, int k, int* __restrict__ out)
{
  __shared__ float tile[128*24];
  const int tid = threadIdx.x;
  const int q   = blockIdx.x*256 + tid;      // all queries in block share batch
  const int b   = q >> 11;

  float fi[24];
  for (int c = 0; c < C; ++c) fi[c] = F[(size_t)q*C + c];

  float topd[24]; int topi[24];
  for (int j = 0; j < 24; ++j){ topd[j] = 3.4e38f; topi[j] = 0; }

  for (int j0 = 0; j0 < NPTS; j0 += 128){
    __syncthreads();
    for (int idx = tid; idx < 128*C; idx += 256)
      tile[idx] = F[((size_t)b*NPTS + j0 + idx/C)*C + (idx % C)];
    __syncthreads();
    for (int jj = 0; jj < 128; ++jj){
      float d = 0.f;
      for (int c = 0; c < C; ++c){ float t = fi[c]-tile[jj*C+c]; d += t*t; }
      if (d < topd[k-1]){
        int pos = k-1;
        while (pos > 0 && d < topd[pos-1]){
          topd[pos]=topd[pos-1]; topi[pos]=topi[pos-1]; --pos;
        }
        topd[pos]=d; topi[pos]=j0+jj;
      }
    }
  }
  for (int j = 0; j < k; ++j) out[(size_t)q*k + j] = topi[j];
}

// e = concat(nb - ctr, ctr) over first 12 channels of X, one batch at a time
__global__ void edge_gather_kernel(const float* __restrict__ X,
                                   const int* __restrict__ kidx,
                                   int b, int k, float* __restrict__ e)
{
  int t = blockIdx.x*256 + threadIdx.x;
  int total = NPTS*k*24;
  if (t >= total) return;
  int c = t % 24, pe = t / 24;
  int j = pe % k, i = pe / k;
  int ri = b*NPTS + i;
  int rj = b*NPTS + kidx[(size_t)ri*k + j];
  float v;
  if (c < 12) v = X[(size_t)rj*24 + c] - X[(size_t)ri*24 + c];
  else        v = X[(size_t)ri*24 + (c-12)];
  e[(size_t)pe*24 + c] = v;
}

__global__ void maxpool_kernel(const float* __restrict__ eh, int k,
                               float* __restrict__ outp)
{
  int t = blockIdx.x*256 + threadIdx.x;   // NPTS*64
  if (t >= NPTS*64) return;
  int c = t & 63, i = t >> 6;
  float m = -3.4e38f;
  for (int j = 0; j < k; ++j)
    m = fmaxf(m, eh[((size_t)(i*k+j))*64 + c]);
  outp[(size_t)i*64 + c] = m;
}

__global__ void reverse_kernel(const float* __restrict__ in,
                               float* __restrict__ out, int C, int M)
{
  int t = blockIdx.x*256 + threadIdx.x;
  if (t >= M*C) return;
  int c = t % C, r = t / C;
  out[(size_t)r*C + c] = in[(size_t)r*C + (C-1-c)];
}

__global__ void scale_bias_kernel(float* __restrict__ x,
                                  const float* __restrict__ logs,
                                  const float* __restrict__ bias,
                                  int inv, int M)
{
  int t = blockIdx.x*256 + threadIdx.x;
  if (t >= M*24) return;
  int c = t % 24;
  float v = x[t];
  x[t] = inv ? (v - bias[c])*expf(-logs[c]) : v*expf(logs[c]) + bias[c];
}

__global__ void ldj0_kernel(float* log_det, const float* __restrict__ logs)
{
  if (threadIdx.x | blockIdx.x) return;
  float s = 0.f;
  for (int c = 0; c < 24; ++c) s += logs[c];
  for (int b = 0; b < NB; ++b) log_det[b] += (float)NPTS * s;
}

// x2' = x2*exp(sc(sv))+tv  (or inverse); ldj += sum(sc(sv)) per batch
__global__ void coupling_kernel(float* __restrict__ x, int ldx, int c1s, int c2s,
                                const float* __restrict__ sv,
                                const float* __restrict__ tv,
                                float* ldj, int M, int inv)
{
  int r = blockIdx.x*256 + threadIdx.x;
  if (r >= M) return;
  int b = r >> 11;
  float ssum = 0.f;
  for (int c = 0; c < c2s; ++c){
    float s = soft_clamp_f(sv[(size_t)r*16 + c]);
    float t = tv[(size_t)r*16 + c];
    float v = x[(size_t)r*ldx + c1s + c];
    x[(size_t)r*ldx + c1s + c] = inv ? (v - t)*expf(-s) : v*expf(s) + t;
    ssum += s;
  }
  if (ldj) atomicAdd(&ldj[b], ssum);
}

__global__ void gauss_acc_kernel(const float* __restrict__ z, int C,
                                 float* acc, int M)
{
  int r = blockIdx.x*256 + threadIdx.x;
  if (r >= M) return;
  int b = r >> 11;
  float s = 0.f;
  for (int c = 0; c < C; ++c){
    float v = z[(size_t)r*C + c];
    s += -0.5f*(v*v + LOG2PI);
  }
  atomicAdd(&acc[b], s);
}

__global__ void concat_kernel(const float* __restrict__ xyz,
                              const float* __restrict__ y,
                              float* __restrict__ x, int M)
{
  int r = blockIdx.x*256 + threadIdx.x;
  if (r >= M) return;
  for (int c = 0; c < 3;  ++c) x[(size_t)r*24 + c]     = xyz[(size_t)r*3 + c];
  for (int c = 0; c < 21; ++c) x[(size_t)r*24 + 3 + c] = y[(size_t)r*21 + c];
}

__global__ void zero_kernel(float* p, int n){
  int t = blockIdx.x*256 + threadIdx.x;
  if (t < n) p[t] = 0.f;
}

__global__ void mask_noise_kernel(float* z, int M){
  int r = blockIdx.x*256 + threadIdx.x;
  if (r >= M) return;
  z[(size_t)r*24+21]=0.f; z[(size_t)r*24+22]=0.f; z[(size_t)r*24+23]=0.f;
}

__global__ void mask_outlier_kernel(float* z, const int* __restrict__ pidx){
  int t = blockIdx.x*256 + threadIdx.x;     // NB*512
  if (t >= NB*512) return;
  int b = t >> 9, j = t & 511;
  size_t r = (size_t)(b*NPTS + pidx[b*NPTS + j])*24;
  z[r+18]=0.f; z[r+19]=0.f; z[r+20]=0.f;
}

// probs = (h @ w + b) / ||w||
__global__ __launch_bounds__(256)
void prob_kernel(const float* __restrict__ h, const float* __restrict__ w,
                 const float* __restrict__ bias, float* __restrict__ probs, int M)
{
  __shared__ float ws[32];
  __shared__ float nrm;
  if (threadIdx.x < 32) ws[threadIdx.x] = w[threadIdx.x];
  __syncthreads();
  if (threadIdx.x == 0){
    float s = 0.f;
    for (int i = 0; i < 32; ++i) s += ws[i]*ws[i];
    nrm = sqrtf(s);
  }
  __syncthreads();
  int r = blockIdx.x*256 + threadIdx.x;
  if (r >= M) return;
  float a = bias[0];
  for (int c = 0; c < 32; ++c) a += h[(size_t)r*32 + c]*ws[c];
  probs[r] = a / nrm;
}

// Descending stable argsort of 2048 keys per batch (bitonic, 1024 threads)
__global__ __launch_bounds__(1024)
void argsort_kernel(const float* __restrict__ probs, int* __restrict__ idx_out)
{
  __shared__ float key[2048];
  __shared__ int   vix[2048];
  const int b = blockIdx.x, tid = threadIdx.x;
  key[tid]      = probs[b*NPTS + tid];      vix[tid]      = tid;
  key[tid+1024] = probs[b*NPTS + tid+1024]; vix[tid+1024] = tid+1024;
  __syncthreads();
  for (int size = 2; size <= 2048; size <<= 1){
    for (int stride = size >> 1; stride > 0; stride >>= 1){
      int i = (tid / stride) * (stride*2) + (tid % stride);
      int j = i + stride;
      bool up = ((i & size) == 0);
      float ki = key[i], kj = key[j];
      int   ii = vix[i], ij = vix[j];
      bool iBeforeJ = (ki > kj) || (ki == kj && ii < ij);   // desc, stable
      bool sw = up ? !iBeforeJ : iBeforeJ;
      if (sw){ key[i]=kj; key[j]=ki; vix[i]=ij; vix[j]=ii; }
      __syncthreads();
    }
  }
  idx_out[b*NPTS + tid]      = vix[tid];
  idx_out[b*NPTS + tid+1024] = vix[tid+1024];
}

__global__ void pool_gather_kernel(const float* __restrict__ x,
                                   const float* __restrict__ probs,
                                   const int* __restrict__ pidx,
                                   float* __restrict__ px)
{
  int t = blockIdx.x*256 + threadIdx.x;     // 12288*24
  if (t >= 12288*24) return;
  int c = t % 24, r2 = t / 24;
  int b = r2 / 1536, i2 = r2 % 1536;
  int ci = pidx[b*NPTS + 512 + i2];
  float yv = 1.f/(1.f + expf(-probs[b*NPTS + ci]));
  px[(size_t)r2*24 + c] = x[((size_t)(b*NPTS + ci))*24 + c]*yv;
}

__global__ void est_kernel(const float* __restrict__ xyz,
                           const int* __restrict__ pidx,
                           const float* __restrict__ d,
                           float* __restrict__ out)
{
  int t = blockIdx.x*256 + threadIdx.x;     // 12288*3
  if (t >= 12288*3) return;
  int c = t % 3, r2 = t / 3;
  int b = r2 / 1536, i2 = r2 % 1536;
  int ci = pidx[b*NPTS + 512 + i2];
  out[(size_t)r2*3 + c] = xyz[((size_t)(b*NPTS + ci))*3 + c] + d[(size_t)r2*4 + c];
}

__global__ void extract_kernel(const float* __restrict__ z, float* __restrict__ out){
  int t = blockIdx.x*256 + threadIdx.x;     // BN*3
  if (t >= BN*3) return;
  int c = t % 3, r = t / 3;
  out[(size_t)r*3 + c] = z[(size_t)r*24 + c];
}

__global__ void finalize_kernel(const float* glz, const float* ld,
                                const float* gle, const float* sldj, float* out)
{
  if (threadIdx.x | blockIdx.x) return;
  float a = 0.f;
  for (int b = 0; b < NB; ++b)
    a += glz[b] + ld[b] - (gle[b] - sldj[b]);
  out[0] = -a / (float)NB;
}

// ---------------------------------------------------------------------------
// Host orchestration
// ---------------------------------------------------------------------------
struct LinP { const float* w; const float* b; };
struct AugP { LinP s0, s1, t0, t1; };
struct FlowP { const float* bias; LinP c1F[3], c1G[3], c2F[3], c2G[3]; const float* logs; };
struct PoolP { LinP d1, d2, d3, l1, l2, l3, prob; };

static const int FLOW_K[12] = {8,16,24,16,16,16,16,16,16,16,16,16};

extern "C" void kernel_launch(void* const* d_in, const int* in_sizes, int n_in,
                              void* d_out, int out_size, void* d_ws, size_t ws_size,
                              hipStream_t stream)
{
  (void)in_sizes; (void)n_in; (void)out_size; (void)ws_size;
  const float* xyz = (const float*)d_in[0];
  const float* eps = (const float*)d_in[1];
  float* out = (float*)d_out;

  // ---- parse params (pytree flatten: sorted dict keys, 'b' before 'w') ----
  int pi = 2;
  auto ARR = [&](){ return (const float*)d_in[pi++]; };
  auto LP  = [&](){ LinP L; L.b = ARR(); L.w = ARR(); return L; };
  AugP aug[3];
  for (int s = 0; s < 3; ++s){ aug[s].s0=LP(); aug[s].s1=LP(); aug[s].t0=LP(); aug[s].t1=LP(); }
  FlowP fl[12];
  for (int i = 0; i < 12; ++i){
    fl[i].bias = ARR();
    for (int j = 0; j < 3; ++j) fl[i].c1F[j] = LP();
    for (int j = 0; j < 3; ++j) fl[i].c1G[j] = LP();
    for (int j = 0; j < 3; ++j) fl[i].c2F[j] = LP();
    for (int j = 0; j < 3; ++j) fl[i].c2G[j] = LP();
    fl[i].logs = ARR();
  }
  PoolP pool;
  pool.d1=LP(); pool.d2=LP(); pool.d3=LP();
  pool.l1=LP(); pool.l2=LP(); pool.l3=LP(); pool.prob=LP();
  LinP sh0 = LP(), sh1 = LP();

  // ---- workspace bump allocator ----
  char* wp = (char*)d_ws;
  auto alloc = [&](size_t bytes){ void* p = wp; wp += (bytes + 255) & ~(size_t)255; return p; };
  float* xA   = (float*)alloc((size_t)BN*24*4);
  float* xB   = (float*)alloc((size_t)BN*24*4);
  float* feat = (float*)alloc((size_t)BN*64*4);
  float* yA   = (float*)alloc((size_t)BN*21*4);
  float* yB   = (float*)alloc((size_t)BN*21*4);
  float* t1   = (float*)alloc((size_t)BN*64*4);
  float* t2   = (float*)alloc((size_t)BN*64*4);
  float* sv   = (float*)alloc((size_t)BN*16*4);
  float* tv   = (float*)alloc((size_t)BN*16*4);
  float* ebuf = (float*)alloc((size_t)NPTS*24*24*4);
  float* ehbuf= (float*)alloc((size_t)NPTS*24*64*4);
  float* hF   = (float*)alloc((size_t)BN*64*4);
  float* hG   = (float*)alloc((size_t)BN*64*4);
  float* pl3  = (float*)alloc((size_t)BN*32*4);
  float* probs= (float*)alloc((size_t)BN*4);
  float* px   = (float*)alloc((size_t)12288*24*4);
  float* pd1  = (float*)alloc((size_t)12288*16*4);
  float* pd2  = (float*)alloc((size_t)12288*16*4);
  float* pd3  = (float*)alloc((size_t)12288*4*4);
  float* accs = (float*)alloc(32*4);
  int*   pidx = (int*)  alloc((size_t)BN*4);
  int*   kidx = (int*)  alloc((size_t)BN*192*4);
  float* gl_eps = accs, *step_ldj = accs+8, *log_det = accs+16, *gl_z = accs+24;

  int koff[12]; { int o = 0; for (int i=0;i<12;++i){ koff[i]=o; o += BN*FLOW_K[i]; } }

  auto LIN = [&](const float* X0,int c0,int ld0, const float* X1,int ld1,
                 LinP L, float* Y,int ldy, int M, int I, int O, int act){
    int nT = (O + 15) >> 4;
    bool simple = (X1 == nullptr) && ((I & 31) == 0) && ((ld0 & 3) == 0) && (c0 >= I);
    if (simple){
      if (nT <= 1)
        lin_wmma_kernel<1,true><<<M/128,256,0,stream>>>(X0,c0,ld0, X1,ld1, L.w,L.b, Y,ldy, M,I,O,act);
      else if (nT == 2)
        lin_wmma_kernel<2,true><<<M/128,256,0,stream>>>(X0,c0,ld0, X1,ld1, L.w,L.b, Y,ldy, M,I,O,act);
      else
        lin_wmma_kernel<4,true><<<M/128,256,0,stream>>>(X0,c0,ld0, X1,ld1, L.w,L.b, Y,ldy, M,I,O,act);
    } else {
      if (nT <= 1)
        lin_wmma_kernel<1,false><<<M/128,256,0,stream>>>(X0,c0,ld0, X1,ld1, L.w,L.b, Y,ldy, M,I,O,act);
      else if (nT == 2)
        lin_wmma_kernel<2,false><<<M/128,256,0,stream>>>(X0,c0,ld0, X1,ld1, L.w,L.b, Y,ldy, M,I,O,act);
      else
        lin_wmma_kernel<4,false><<<M/128,256,0,stream>>>(X0,c0,ld0, X1,ld1, L.w,L.b, Y,ldy, M,I,O,act);
    }
  };
  auto MLP3 = [&](LinP* ps, const float* xin, float* outb){   // 12->64->64->12 over BN
    LIN(xin,12,24, nullptr,0, ps[0], t1,64, BN,12,64, ACT_LRELU);
    LIN(t1 ,64,64, nullptr,0, ps[1], t2,64, BN,64,64, ACT_LRELU);
    LIN(t2 ,64,64, nullptr,0, ps[2], outb,16, BN,64,12, ACT_NONE);
  };
  auto KNNCONV = [&](LinP* F, LinP* G, const float* xin, const int* ki, int k,
                     float* svb, float* tvb){
    int Me = NPTS*k;
    for (int b = 0; b < NB; ++b){
      int totE = NPTS*k*24;
      edge_gather_kernel<<<(totE+255)/256,256,0,stream>>>(xin, ki, b, k, ebuf);
      LIN(ebuf,24,24, nullptr,0, F[0], ehbuf,64, Me,24,64, ACT_LRELU);
      maxpool_kernel<<<(NPTS*64)/256,256,0,stream>>>(ehbuf, k, hF + (size_t)b*NPTS*64);
      LIN(ebuf,24,24, nullptr,0, G[0], ehbuf,64, Me,24,64, ACT_LRELU);
      maxpool_kernel<<<(NPTS*64)/256,256,0,stream>>>(ehbuf, k, hG + (size_t)b*NPTS*64);
    }
    LIN(hF,64,64, nullptr,0, F[1], t1,64, BN,64,64, ACT_LRELU);
    LIN(t1,64,64, nullptr,0, F[2], svb,16, BN,64,12, ACT_NONE);
    LIN(hG,64,64, nullptr,0, G[1], t2,64, BN,64,64, ACT_LRELU);
    LIN(t2,64,64, nullptr,0, G[2], tvb,16, BN,64,12, ACT_NONE);
  };
  auto REV = [&](float*& a, float*& b, int C){
    reverse_kernel<<<(BN*C)/256,256,0,stream>>>(a, b, C, BN);
    float* t = a; a = b; b = t;
  };

  // ---- accumulators & base log-probs ----
  zero_kernel<<<1,32,0,stream>>>(accs, 32);
  gauss_acc_kernel<<<BN/256,256,0,stream>>>(eps, 21, gl_eps, BN);

  // ---- shallow feature net: 3 -> 32 -> 64 (relu) ----
  LIN(xyz,3,3, nullptr,0, sh0, t1,32, BN,3,32, ACT_RELU);
  LIN(t1 ,32,32, nullptr,0, sh1, feat,64, BN,32,64, ACT_RELU);

  // ---- augmentation steps on y (21 ch) ----
  hipMemcpyAsync(yA, eps, (size_t)BN*21*4, hipMemcpyDeviceToDevice, stream);
  const bool AUGR[3] = {false, true, false};
  for (int s = 0; s < 3; ++s){
    if (AUGR[s]) REV(yA, yB, 21);
    LIN(yA,11,21, feat,64, aug[s].s0, t1,64, BN,75,64, ACT_LRELU);
    LIN(t1 ,64,64, nullptr,0, aug[s].s1, sv,16, BN,64,10, ACT_NONE);
    LIN(yA,11,21, feat,64, aug[s].t0, t2,64, BN,75,64, ACT_LRELU);
    LIN(t2 ,64,64, nullptr,0, aug[s].t1, tv,16, BN,64,10, ACT_NONE);
    coupling_kernel<<<BN/256,256,0,stream>>>(yA,21,11,10, sv,tv, step_ldj, BN, 0);
    if (AUGR[s]) REV(yA, yB, 21);
  }

  // ---- x = concat(xyz, y) ----
  concat_kernel<<<BN/256,256,0,stream>>>(xyz, yA, xA, BN);

  // ---- forward flows ----
  for (int i = 0; i < 12; ++i){
    int k = FLOW_K[i];
    int* ki = kidx + koff[i];
    const float* kf = (i < 3) ? xyz : xA;
    int kC = (i < 3) ? 3 : 24;
    knn_kernel<<<BN/256,256,0,stream>>>(kf, kC, k, ki);

    REV(xA, xB, 24);
    scale_bias_kernel<<<(BN*24)/256,256,0,stream>>>(xA, fl[i].logs, fl[i].bias, 0, BN);
    ldj0_kernel<<<1,1,0,stream>>>(log_det, fl[i].logs);

    MLP3(fl[i].c1F, xA, sv);
    MLP3(fl[i].c1G, xA, tv);
    coupling_kernel<<<BN/256,256,0,stream>>>(xA,24,12,12, sv,tv, log_det, BN, 0);

    REV(xA, xB, 24);
    KNNCONV(fl[i].c2F, fl[i].c2G, xA, ki, k, sv, tv);
    coupling_kernel<<<BN/256,256,0,stream>>>(xA,24,12,12, sv,tv, log_det, BN, 0);

    if (i == 2){
      // outlier pool on (xyz, x)
      LIN(xA,24,24, xyz,3, pool.l1, t1,64, BN,27,64, ACT_RELU);
      LIN(t1 ,64,64, nullptr,0, pool.l2, t2,64, BN,64,64, ACT_RELU);
      LIN(t2 ,64,64, nullptr,0, pool.l3, pl3,32, BN,64,32, ACT_RELU);
      prob_kernel<<<BN/256,256,0,stream>>>(pl3, pool.prob.w, pool.prob.b, probs, BN);
      argsort_kernel<<<NB,1024,0,stream>>>(probs, pidx);
      pool_gather_kernel<<<(12288*24)/256,256,0,stream>>>(xA, probs, pidx, px);
      LIN(px ,24,24, nullptr,0, pool.d1, pd1,16, 12288,24,12, ACT_RELU);
      LIN(pd1,12,16, nullptr,0, pool.d2, pd2,16, 12288,12,6,  ACT_RELU);
      LIN(pd2, 6,16, nullptr,0, pool.d3, pd3, 4, 12288, 6,3,  ACT_NONE);
      est_kernel<<<(12288*3+255)/256,256,0,stream>>>(xyz, pidx, pd3, out + 49153);
    }
  }

  // ---- total loss ----
  gauss_acc_kernel<<<BN/256,256,0,stream>>>(xA, 24, gl_z, BN);
  finalize_kernel<<<1,1,0,stream>>>(gl_z, log_det, gl_eps, step_ldj, out + 49152);

  // ---- mask noise / outlier channels ----
  mask_noise_kernel<<<BN/256,256,0,stream>>>(xA, BN);
  mask_outlier_kernel<<<(NB*512)/256,256,0,stream>>>(xA, pidx);

  // ---- inverse flows ----
  for (int i = 11; i >= 0; --i){
    int k = FLOW_K[i];
    int* ki = kidx + koff[i];
    KNNCONV(fl[i].c2F, fl[i].c2G, xA, ki, k, sv, tv);
    coupling_kernel<<<BN/256,256,0,stream>>>(xA,24,12,12, sv,tv, nullptr, BN, 1);
    REV(xA, xB, 24);
    MLP3(fl[i].c1F, xA, sv);
    MLP3(fl[i].c1G, xA, tv);
    coupling_kernel<<<BN/256,256,0,stream>>>(xA,24,12,12, sv,tv, nullptr, BN, 1);
    scale_bias_kernel<<<(BN*24)/256,256,0,stream>>>(xA, fl[i].logs, fl[i].bias, 1, BN);
    REV(xA, xB, 24);
  }

  // ---- output denoised xyz ----
  extract_kernel<<<(BN*3)/256,256,0,stream>>>(xA, out);
}
